// GNNChildDecoder_26577257628370
// MI455X (gfx1250) — matrix operand order repaired and checked by reading, not compiled, not run on hardware
//
#include <hip/hip_runtime.h>
#include <hip/hip_bf16.h>

// ---------------------------------------------------------------------------
// GNNChildDecoder for MI455X (gfx1250, wave32, WMMA, async LDS copies).
// Algebraic restructurings: pairwise-concat GEMMs become rank-separated
// A[i]+B[j] forms; edge MLP splits into per-node GEMMs + one big elat@W3 GEMM
// whose masked-relu/segment-max epilogue is fused into the GEMM tile (atomic
// float-max via uint atomicMax on non-negative values), eliminating the 64 MB
// R intermediate. GEMMs use v_wmma_f32_16x16x32_bf16 with double-buffered
// global_load_async_to_lds_b128 staging throttled by s_wait_asynccnt.
// ---------------------------------------------------------------------------

typedef __attribute__((ext_vector_type(16))) __bf16 v16bf;
typedef __attribute__((ext_vector_type(8)))  float  v8f;

struct alignas(16) U4 { unsigned int x, y, z, w; };
union FragU { U4 u[2]; v16bf v; };

__device__ inline unsigned short f2bf(float f) {
  unsigned u = __float_as_uint(f);
  unsigned r = (u + 0x7fffu + ((u >> 16) & 1u)) >> 16;   // round-nearest-even
  return (unsigned short)r;
}

// Low 32 bits of a generic pointer into LDS == wave-relative LDS byte address
// (ISA 10.2: flat->LDS mapping truncates to addr[31:0]).
__device__ inline unsigned ldsaddr(const void* p) {
  return (unsigned)(unsigned long long)p;
}

// Async direct global->LDS 16B copy (VGLOBAL op 98, tracked by ASYNCcnt).
__device__ inline void async_copy_b128(unsigned lds_off, const void* gaddr) {
  asm volatile("global_load_async_to_lds_b128 %0, %1, off"
               :: "v"(lds_off), "v"(gaddr)
               : "memory");
}

#define TM 64
#define TN 64
#define TKH 32      // K halves per step
#define LSTR 40     // padded LDS row stride (halves) to dodge bank conflicts

// ---------------------------------------------------------------------------
// Generic bf16 GEMM: C[MxN] (f32) = act(A[MxK] @ B[KxN] + bias)
// A row-major bf16; BT is B pre-transposed, [N][K] row-major bf16.
// Block = 128 threads (4 waves); tile 64x64, K-step 32; double-buffered LDS
// filled by async copies so the next tile streams while WMMAs run.
// Fragment layouts per CDNA5 ISA 7.12.2:
//   A (16-bit 16x32): lane<16 row=l, halves = K[0..7],K[16..23]; lane>=16 +8.
//   B (16-bit 32x16): lane<16 col=l, halves = K[0..7],K[16..23]; lane>=16 +8.
// ---------------------------------------------------------------------------
__global__ void __launch_bounds__(128)
gemm_bf16_kernel(const unsigned short* __restrict__ A,
                 const unsigned short* __restrict__ BT,
                 float* __restrict__ C,
                 int M, int N, int K,
                 const float* __restrict__ bias, int do_relu)
{
  __shared__ unsigned short As[2][TM * LSTR];
  __shared__ unsigned short Bs[2][TN * LSTR];

  const int tid  = threadIdx.x;
  const int lane = tid & 31;
  const int wave = tid >> 5;
  const int m0 = blockIdx.x * TM;
  const int n0 = blockIdx.y * TN;

  const int ra0 = tid >> 2,         oa0 = (tid & 3) * 8;
  const int ra1 = (tid + 128) >> 2, oa1 = ((tid + 128) & 3) * 8;

  v8f acc[4];
#pragma unroll
  for (int s = 0; s < 4; ++s)
#pragma unroll
    for (int e = 0; e < 8; ++e) acc[s][e] = 0.0f;

  auto stage = [&](int buf, int kk) {
    async_copy_b128(ldsaddr(&As[buf][ra0 * LSTR + oa0]),
                    &A[(size_t)(m0 + ra0) * K + kk + oa0]);
    async_copy_b128(ldsaddr(&As[buf][ra1 * LSTR + oa1]),
                    &A[(size_t)(m0 + ra1) * K + kk + oa1]);
    async_copy_b128(ldsaddr(&Bs[buf][ra0 * LSTR + oa0]),
                    &BT[(size_t)(n0 + ra0) * K + kk + oa0]);
    async_copy_b128(ldsaddr(&Bs[buf][ra1 * LSTR + oa1]),
                    &BT[(size_t)(n0 + ra1) * K + kk + oa1]);
  };

  stage(0, 0);
  const int nt = K / TKH;

  const int ar = wave * 16 + (lane & 15);
  const int ks = (lane & 16) ? 8 : 0;
  const int kb = (lane & 16) ? 8 : 0;

  for (int t = 0; t < nt; ++t) {
    const int buf = t & 1;
    if (t + 1 < nt) {
      stage(buf ^ 1, (t + 1) * TKH);
      asm volatile("s_wait_asynccnt 0x4" ::: "memory");
    } else {
      asm volatile("s_wait_asynccnt 0x0" ::: "memory");
    }
    __syncthreads();

    FragU fa, fb[4];
    fa.u[0] = *(const U4*)&As[buf][ar * LSTR + ks];
    fa.u[1] = *(const U4*)&As[buf][ar * LSTR + ks + 16];
#pragma unroll
    for (int s = 0; s < 4; ++s) {
      const int br = s * 16 + (lane & 15);
      fb[s].u[0] = *(const U4*)&Bs[buf][br * LSTR + kb];
      fb[s].u[1] = *(const U4*)&Bs[buf][br * LSTR + kb + 16];
    }
#pragma unroll
    for (int s = 0; s < 4; ++s)
      acc[s] = __builtin_amdgcn_wmma_f32_16x16x32_bf16(
          false, fa.v, false, fb[s].v, (short)0, acc[s], false, false);
    __syncthreads();
  }

  const int rbase = m0 + wave * 16 + ((lane & 16) ? 8 : 0);
#pragma unroll
  for (int s = 0; s < 4; ++s) {
    const int c = n0 + s * 16 + (lane & 15);
    const float bv = bias ? bias[c] : 0.0f;
#pragma unroll
    for (int v = 0; v < 8; ++v) {
      float x = acc[s][v] + bv;
      if (do_relu) x = fmaxf(x, 0.0f);
      C[(size_t)(rbase + v) * N + c] = x;
    }
  }
}

// ---------------------------------------------------------------------------
// Fused R-GEMM + edge message + segment_max:
//   R = elat @ W3 (same WMMA pipeline), then per D-tile element (j-row, h-col):
//   cand(t) = relu(R + P[i,h]+bop[h] + Q[j,h] + eel[i,j,t]*W4[t,h]) if
//             eel>0 & exists[i]>0 & exists[j]>0, else 0.
//   cf[i,h] = max over j,t (atomicMax; cf pre-zeroed; all candidates >= 0 so
//   uint bit-pattern atomicMax == float max, order-independent/deterministic).
// A 64-row M-tile spans exactly one source node i (64 | 256).
// ---------------------------------------------------------------------------
__global__ void __launch_bounds__(128)
gemm_edge_kernel(const unsigned short* __restrict__ A,    // elat bf16 [65536x256]
                 const unsigned short* __restrict__ BT,   // W3T bf16 [256x256]
                 const float* __restrict__ P, const float* __restrict__ Q,
                 const float* __restrict__ eel, const float* __restrict__ exists,
                 const float* __restrict__ Wop_it, const float* __restrict__ bop_it,
                 float* __restrict__ cf)                   // [256x256] zeroed
{
  __shared__ unsigned short As[2][TM * LSTR];
  __shared__ unsigned short Bs[2][TN * LSTR];
  __shared__ float eelS[128];              // (64 j) x (2 t)
  __shared__ float exS[64];
  __shared__ float PbS[64], w40S[64], w41S[64];

  const int tid  = threadIdx.x;
  const int lane = tid & 31;
  const int wave = tid >> 5;
  const int m0 = blockIdx.x * TM;          // elat row base
  const int n0 = blockIdx.y * TN;
  const int K  = 256;
  const int i  = m0 >> 8;                  // source node of this whole tile
  const int jb = m0 & 255;                 // first j in tile

  eelS[tid] = eel[(size_t)(i * 256 + jb) * 2 + tid];
  if (tid < 64) {
    exS[tid]  = exists[jb + tid];
    PbS[tid]  = P[i * 256 + n0 + tid] + bop_it[n0 + tid];
    w40S[tid] = Wop_it[768 * 256 + n0 + tid];
    w41S[tid] = Wop_it[769 * 256 + n0 + tid];
  }

  const int ra0 = tid >> 2,         oa0 = (tid & 3) * 8;
  const int ra1 = (tid + 128) >> 2, oa1 = ((tid + 128) & 3) * 8;

  v8f acc[4];
#pragma unroll
  for (int s = 0; s < 4; ++s)
#pragma unroll
    for (int e = 0; e < 8; ++e) acc[s][e] = 0.0f;

  auto stage = [&](int buf, int kk) {
    async_copy_b128(ldsaddr(&As[buf][ra0 * LSTR + oa0]),
                    &A[(size_t)(m0 + ra0) * K + kk + oa0]);
    async_copy_b128(ldsaddr(&As[buf][ra1 * LSTR + oa1]),
                    &A[(size_t)(m0 + ra1) * K + kk + oa1]);
    async_copy_b128(ldsaddr(&Bs[buf][ra0 * LSTR + oa0]),
                    &BT[(size_t)(n0 + ra0) * K + kk + oa0]);
    async_copy_b128(ldsaddr(&Bs[buf][ra1 * LSTR + oa1]),
                    &BT[(size_t)(n0 + ra1) * K + kk + oa1]);
  };

  stage(0, 0);
  const int ar = wave * 16 + (lane & 15);
  const int ks = (lane & 16) ? 8 : 0;
  const int kb = (lane & 16) ? 8 : 0;

  for (int t = 0; t < 8; ++t) {           // 256/32 K-steps
    const int buf = t & 1;
    if (t + 1 < 8) {
      stage(buf ^ 1, (t + 1) * TKH);
      asm volatile("s_wait_asynccnt 0x4" ::: "memory");
    } else {
      asm volatile("s_wait_asynccnt 0x0" ::: "memory");
    }
    __syncthreads();

    FragU fa, fb[4];
    fa.u[0] = *(const U4*)&As[buf][ar * LSTR + ks];
    fa.u[1] = *(const U4*)&As[buf][ar * LSTR + ks + 16];
#pragma unroll
    for (int s = 0; s < 4; ++s) {
      const int br = s * 16 + (lane & 15);
      fb[s].u[0] = *(const U4*)&Bs[buf][br * LSTR + kb];
      fb[s].u[1] = *(const U4*)&Bs[buf][br * LSTR + kb + 16];
    }
#pragma unroll
    for (int s = 0; s < 4; ++s)
      acc[s] = __builtin_amdgcn_wmma_f32_16x16x32_bf16(
          false, fa.v, false, fb[s].v, (short)0, acc[s], false, false);
    __syncthreads();
  }

  // Fused epilogue: masked relu + max over this tile's 16 j-rows per column.
  float mcol[4] = {0.0f, 0.0f, 0.0f, 0.0f};
  if (exists[i] > 0.0f) {
    const int jl_base = wave * 16 + ((lane & 16) ? 8 : 0);
#pragma unroll
    for (int v = 0; v < 8; ++v) {
      const int jl = jl_base + v;                 // 0..63
      const float l0 = eelS[2 * jl], l1 = eelS[2 * jl + 1];
      const bool ej = exS[jl] > 0.0f;
      if (!ej || (l0 <= 0.0f && l1 <= 0.0f)) continue;
      const int j = jb + jl;
#pragma unroll
      for (int s = 0; s < 4; ++s) {
        const int cl = s * 16 + (lane & 15);
        const float val = acc[s][v] + PbS[cl] + Q[j * 256 + n0 + cl];
        if (l0 > 0.0f) mcol[s] = fmaxf(mcol[s], fmaxf(fmaf(l0, w40S[cl], val), 0.0f));
        if (l1 > 0.0f) mcol[s] = fmaxf(mcol[s], fmaxf(fmaf(l1, w41S[cl], val), 0.0f));
      }
    }
  }
#pragma unroll
  for (int s = 0; s < 4; ++s) {
    const float m2 = fmaxf(mcol[s], __shfl_xor(mcol[s], 16));
    if (lane < 16 && m2 > 0.0f)
      atomicMax((unsigned int*)&cf[i * 256 + n0 + s * 16 + lane],
                __float_as_uint(m2));
  }
}

// ---------------------------------------------------------------------------
__global__ void __launch_bounds__(256)
zero_kernel(float* __restrict__ p)
{
  p[blockIdx.x * 256 + threadIdx.x] = 0.0f;
}

// child = relu(parent @ Wp + bp); streams the 64 MB Wp once, coalesced.
__global__ void __launch_bounds__(256)
child_kernel(const float* __restrict__ parent, const float* __restrict__ Wp,
             const float* __restrict__ bp, float* __restrict__ child,
             unsigned short* __restrict__ cfbf, unsigned short* __restrict__ allf)
{
  __shared__ float pl[256];
  const int tid = threadIdx.x;
  pl[tid] = parent[tid];
  __syncthreads();
  const int c = blockIdx.x * 256 + tid;
  float s = bp[c];
  for (int f = 0; f < 256; ++f) s = fmaf(pl[f], Wp[(size_t)f * 65536 + c], s);
  s = fmaxf(s, 0.0f);
  child[c] = s;
  const unsigned short b = f2bf(s);
  cfbf[c] = b;
  allf[(c >> 8) * 768 + (c & 255)] = b;
}

__global__ void __launch_bounds__(256)
exists_kernel(const float* __restrict__ child, const float* __restrict__ We,
              const float* __restrict__ be, float* __restrict__ out_exists)
{
  const int i = threadIdx.x;
  float s = be[0];
  for (int h = 0; h < 256; ++h) s = fmaf(child[i * 256 + h], We[h], s);
  out_exists[i] = s;
}

// elat[i,j,:] = relu(A[i,:] + B[j,:] + bel) stored bf16;
// eel[i,j,t]  = dot(elat[i,j,:], Wee[t,:]) + bee[t] in f32 (it is an output).
__global__ void __launch_bounds__(32)
elat_kernel(const float* __restrict__ Amat, const float* __restrict__ Bmat,
            const float* __restrict__ bel, const float* __restrict__ Wee,
            const float* __restrict__ bee, unsigned short* __restrict__ elat,
            float* __restrict__ eel_out)
{
  const int ij = blockIdx.x;
  const int i = ij >> 8, j = ij & 255;
  const int lane = threadIdx.x;
  float a0 = 0.0f, a1 = 0.0f;
#pragma unroll
  for (int u = 0; u < 8; ++u) {
    const int h = lane + u * 32;
    float e = Amat[i * 256 + h] + Bmat[j * 256 + h] + bel[h];
    e = fmaxf(e, 0.0f);
    elat[(size_t)ij * 256 + h] = f2bf(e);
    a0 = fmaf(e, Wee[h], a0);
    a1 = fmaf(e, Wee[256 + h], a1);
  }
#pragma unroll
  for (int o = 16; o > 0; o >>= 1) {
    a0 += __shfl_down(a0, o);
    a1 += __shfl_down(a1, o);
  }
  if (lane == 0) {
    eel_out[ij * 2 + 0] = a0 + bee[0];
    eel_out[ij * 2 + 1] = a1 + bee[1];
  }
}

// f32 -> bf16 convert, writing up to two strided destinations.
__global__ void __launch_bounds__(256)
convert_kernel(const float* __restrict__ src,
               unsigned short* __restrict__ dst1, int stride1, int ofs1,
               unsigned short* __restrict__ dst2, int stride2, int ofs2, int cols)
{
  const int idx = blockIdx.x * 256 + threadIdx.x;
  const int r = idx / cols, c = idx % cols;
  const unsigned short b = f2bf(src[idx]);
  dst1[r * stride1 + ofs1 + c] = b;
  if (dst2) dst2[r * stride2 + ofs2 + c] = b;
}

// Transpose + convert: dst_bf16[n*K + k] = src_f32[k*ld + n].
__global__ void __launch_bounds__(256)
transpose_bf16_kernel(const float* __restrict__ src, int ld,
                      unsigned short* __restrict__ dst, int K, int N)
{
  const int k = blockIdx.x * 16 + threadIdx.x;
  const int n = blockIdx.y * 16 + threadIdx.y;
  if (k < K && n < N) dst[(size_t)n * K + k] = f2bf(src[(size_t)k * ld + n]);
}

// ---------------------------------------------------------------------------
extern "C" void kernel_launch(void* const* d_in, const int* in_sizes, int n_in,
                              void* d_out, int out_size, void* d_ws, size_t ws_size,
                              hipStream_t stream)
{
  const float* parent = (const float*)d_in[0];
  const float* Wp  = (const float*)d_in[1];
  const float* bp  = (const float*)d_in[2];
  const float* We  = (const float*)d_in[3];
  const float* be  = (const float*)d_in[4];
  const float* Wel = (const float*)d_in[5];
  const float* bel = (const float*)d_in[6];
  const float* Wee = (const float*)d_in[7];
  const float* bee = (const float*)d_in[8];
  const float* Wop = (const float*)d_in[9];
  const float* bop = (const float*)d_in[10];
  const float* Wc  = (const float*)d_in[11];
  const float* bc  = (const float*)d_in[12];
  const float* Wsm = (const float*)d_in[13];
  const float* bs  = (const float*)d_in[14];
  const float* Wc2 = (const float*)d_in[15];
  const float* bc2 = (const float*)d_in[16];

  // Output layout (floats): out_child[65536] | sem[16384] | exists[256] | eel[131072]
  float* out   = (float*)d_out;
  float* outc  = out;
  float* sem   = out + 65536;
  float* exis  = out + 81920;
  float* eel   = out + 82176;

  char* wsp = (char*)d_ws;
  auto carve = [&](size_t bytes) -> void* {
    void* r = (void*)wsp;
    wsp += (bytes + 255) & ~(size_t)255;
    return r;
  };
  float* child = (float*)carve(65536 * 4);
  float* Amat  = (float*)carve(65536 * 4);
  float* Bmat  = (float*)carve(65536 * 4);
  float* Pm    = (float*)carve(65536 * 4);
  float* Qm    = (float*)carve(65536 * 4);
  float* cf    = (float*)carve(65536 * 4);
  float* cf2   = (float*)carve(65536 * 4);
  unsigned short* cfbf    = (unsigned short*)carve(65536 * 2);
  unsigned short* cf2bf   = (unsigned short*)carve(65536 * 2);
  unsigned short* allfbf  = (unsigned short*)carve(196608 * 2);
  unsigned short* WelTopT = (unsigned short*)carve(65536 * 2);
  unsigned short* WelBotT = (unsigned short*)carve(65536 * 2);
  unsigned short* W1T     = (unsigned short*)carve(65536 * 2);
  unsigned short* W2T     = (unsigned short*)carve(65536 * 2);
  unsigned short* W3T     = (unsigned short*)carve(65536 * 2);
  unsigned short* WcT     = (unsigned short*)carve(196608 * 2);
  unsigned short* WsT     = (unsigned short*)carve(16384 * 2);
  unsigned short* Wc2T    = (unsigned short*)carve(65536 * 2);
  unsigned short* elatbf  = (unsigned short*)carve((size_t)16777216 * 2); // 32 MB

  const dim3 tb(16, 16);
  transpose_bf16_kernel<<<dim3(16, 16), tb, 0, stream>>>(Wel,             256, WelTopT, 256, 256);
  transpose_bf16_kernel<<<dim3(16, 16), tb, 0, stream>>>(Wel + 256 * 256, 256, WelBotT, 256, 256);
  transpose_bf16_kernel<<<dim3(48, 16), tb, 0, stream>>>(Wc,              256, WcT,     768, 256);
  transpose_bf16_kernel<<<dim3(16, 4),  tb, 0, stream>>>(Wsm,              64, WsT,     256, 64);
  transpose_bf16_kernel<<<dim3(16, 16), tb, 0, stream>>>(Wc2,             256, Wc2T,    256, 256);

  child_kernel<<<256, 256, 0, stream>>>(parent, Wp, bp, child, cfbf, allfbf);
  exists_kernel<<<1, 256, 0, stream>>>(child, We, be, exis);

  gemm_bf16_kernel<<<dim3(4, 4), 128, 0, stream>>>(cfbf, WelTopT, Amat, 256, 256, 256, nullptr, 0);
  gemm_bf16_kernel<<<dim3(4, 4), 128, 0, stream>>>(cfbf, WelBotT, Bmat, 256, 256, 256, nullptr, 0);

  elat_kernel<<<65536, 32, 0, stream>>>(Amat, Bmat, bel, Wee, bee, elatbf, eel);

  for (int it = 0; it < 2; ++it) {
    const float* Wop_it = Wop + (size_t)it * 770 * 256;
    transpose_bf16_kernel<<<dim3(16, 16), tb, 0, stream>>>(Wop_it,             256, W1T, 256, 256);
    transpose_bf16_kernel<<<dim3(16, 16), tb, 0, stream>>>(Wop_it + 256 * 256, 256, W2T, 256, 256);
    transpose_bf16_kernel<<<dim3(16, 16), tb, 0, stream>>>(Wop_it + 512 * 256, 256, W3T, 256, 256);

    gemm_bf16_kernel<<<dim3(4, 4), 128, 0, stream>>>(cfbf, W1T, Pm, 256, 256, 256, nullptr, 0);
    gemm_bf16_kernel<<<dim3(4, 4), 128, 0, stream>>>(cfbf, W2T, Qm, 256, 256, 256, nullptr, 0);

    zero_kernel<<<256, 256, 0, stream>>>(cf);
    gemm_edge_kernel<<<dim3(1024, 4), 128, 0, stream>>>(
        elatbf, W3T, Pm, Qm, eel, exis, Wop_it, bop + it * 256, cf);

    convert_kernel<<<256, 256, 0, stream>>>(cf, cfbf, 256, 0, allfbf, 768, 256 * (it + 1), 256);
  }

  gemm_bf16_kernel<<<dim3(4, 4), 128, 0, stream>>>(allfbf, WcT, cf2, 256, 256, 768, bc, 1);
  convert_kernel<<<256, 256, 0, stream>>>(cf2, cf2bf, 256, 0, nullptr, 0, 0, 256);
  gemm_bf16_kernel<<<dim3(4, 1), 128, 0, stream>>>(cf2bf, WsT,  sem,  256, 64,  256, bs,  0);
  gemm_bf16_kernel<<<dim3(4, 4), 128, 0, stream>>>(cf2bf, Wc2T, outc, 256, 256, 256, bc2, 1);
}
